// CGDN_74637941670221
// MI455X (gfx1250) — matrix operand order repaired and checked by reading, not compiled, not run on hardware
//
#include <hip/hip_runtime.h>
#include <hip/hip_bf16.h>
#include <math.h>

typedef __attribute__((ext_vector_type(16))) __bf16 v16bf;
typedef __attribute__((ext_vector_type(8)))  __bf16 v8bf;
typedef __attribute__((ext_vector_type(8)))  float  v8f;

#define HID   128
#define HEADS 4
#define CH    32
#define LAY   4

__device__ __forceinline__ float gelu_erf(float x) {
    return 0.5f * x * (1.0f + erff(x * 0.70710678118654752440f));
}
// monotone float -> uint key so integer atomicMax implements float segment-max
__device__ __forceinline__ unsigned fkey(float f) {
    unsigned u = __float_as_uint(f);
    return (u & 0x80000000u) ? ~u : (u | 0x80000000u);
}
__device__ __forceinline__ float funkey(unsigned k) {
    unsigned u = (k & 0x80000000u) ? (k ^ 0x80000000u) : ~k;
    return __uint_as_float(u);
}

// ---------------------------------------------------------------------------
// Pack Wl/Wr (fp32 [L,128,128] row-major, W[k][n]) into bf16 per-lane WMMA B
// layout: out[((((l*2+mat)*4+kc)*8+nt)*32+lane)*16 + e]
//   lane<16 : k = kc*32 +      e, n = nt*16 + lane
//   lane>=16: k = kc*32 + 16 + e, n = nt*16 + (lane-16)
// ---------------------------------------------------------------------------
__global__ void k_prep_w(const float* __restrict__ Wl, const float* __restrict__ Wr,
                         __bf16* __restrict__ wpack) {
    int tid  = blockIdx.x * blockDim.x + threadIdx.x;   // 131072 total
    int e    = tid & 15;
    int lane = (tid >> 4) & 31;
    int nt   = (tid >> 9) & 7;
    int kc   = (tid >> 12) & 3;
    int mat  = (tid >> 14) & 1;
    int l    = tid >> 15;
    const float* W = mat ? Wr : Wl;
    int k = kc * 32 + ((lane >= 16) ? 16 : 0) + e;
    int n = nt * 16 + (lane & 15);
    wpack[tid] = (__bf16)W[l * HID * HID + k * HID + n];
}

// ---------------------------------------------------------------------------
// Encoder: h = GELU(LN(x @ enc_W + enc_b)); writes fp32 h and bf16 h
// ---------------------------------------------------------------------------
__global__ void k_encoder(const float* __restrict__ x, const float* __restrict__ W,
                          const float* __restrict__ b, const float* __restrict__ g,
                          const float* __restrict__ be,
                          float* __restrict__ h, __bf16* __restrict__ hbf) {
    int n = blockIdx.x, t = threadIdx.x;
    __shared__ float xr[6];
    __shared__ float red[HID];
    if (t < 6) xr[t] = x[n * 6 + t];
    __syncthreads();
    float v = b[t];
    #pragma unroll
    for (int j = 0; j < 6; ++j) v = fmaf(xr[j], W[j * HID + t], v);
    red[t] = v; __syncthreads();
    for (int s2 = 64; s2 > 0; s2 >>= 1) { if (t < s2) red[t] += red[t + s2]; __syncthreads(); }
    float mu = red[0] * (1.0f / HID); __syncthreads();
    float d = v - mu;
    red[t] = d * d; __syncthreads();
    for (int s2 = 64; s2 > 0; s2 >>= 1) { if (t < s2) red[t] += red[t + s2]; __syncthreads(); }
    float var = red[0] * (1.0f / HID);
    float y = gelu_erf(d * rsqrtf(var + 1e-5f) * g[t] + be[t]);
    h[n * HID + t]   = y;
    hbf[n * HID + t] = (__bf16)y;
}

// ---------------------------------------------------------------------------
// FiLM: film = GELU(target@W1+b1)@W2+b2 ; gamma=film[:128], beta=film[128:]
// ---------------------------------------------------------------------------
__global__ void k_film(const float* __restrict__ target,
                       const float* __restrict__ W1, const float* __restrict__ b1,
                       const float* __restrict__ W2, const float* __restrict__ b2,
                       float* __restrict__ gamma, float* __restrict__ beta) {
    int n = blockIdx.x, t = threadIdx.x;
    __shared__ float t1[64];
    if (t < 64) t1[t] = gelu_erf(target[n] * W1[t] + b1[t]);
    __syncthreads();
    float v = b2[t];
    #pragma unroll 8
    for (int j = 0; j < 64; ++j) v = fmaf(t1[j], W2[j * 256 + t], v);
    if (t < HID) gamma[n * HID + t] = v;
    else         beta[n * HID + (t - HID)] = v;
}

// ---------------------------------------------------------------------------
// WMMA GEMM: xl = h@Wl[l]+bl[l], xr = h@Wr[l]+br[l]
// One wave computes a 16x32 output slab (two adjacent 16x16 tiles) so the A
// fragment is loaded once and fed to two v_wmma streams per K-chunk.
// 8 x v_wmma_f32_16x16x32_bf16 per wave (K=128, 2 n-tiles).
// ---------------------------------------------------------------------------
__global__ void k_gemm_xlxr(const __bf16* __restrict__ hbf,
                            const __bf16* __restrict__ wpack,
                            const float* __restrict__ bl, const float* __restrict__ br,
                            float* __restrict__ xl, float* __restrict__ xr,
                            int layer, int N) {
    int gwave = (blockIdx.x * blockDim.x + threadIdx.x) >> 5;
    int lane  = threadIdx.x & 31;
    int rowTile = gwave >> 3;        // 8 waves per 16-row strip: 4 nt-pairs x 2 mats
    int sub  = gwave & 7;
    int mat  = sub >> 2;
    int nt0  = (sub & 3) * 2;        // this wave: column tiles nt0, nt0+1
    int m    = lane & 15;
    int hi   = lane >> 4;            // lane half selects A K-offset / C row group
    int koff = hi ? 16 : 0;

    const __bf16* arow  = hbf + (size_t)(rowTile * 16 + m) * HID;
    const __bf16* bbase = wpack +
        (size_t)((((layer * 2 + mat) * 4 + 0) * 8 + nt0) * 32 + lane) * 16;

    v8f acc0 = {};
    v8f acc1 = {};
    #pragma unroll
    for (int kc = 0; kc < 4; ++kc) {
        union { v16bf v; v8bf h2[2]; } A;
        A.h2[0] = *(const v8bf*)(arow + kc * 32 + koff);        // K = kc*32 + koff..+7
        A.h2[1] = *(const v8bf*)(arow + kc * 32 + koff + 8);    // K = kc*32 + koff+8..+15
        v16bf B0 = *(const v16bf*)(bbase + kc * 4096);          // tile nt0
        v16bf B1 = *(const v16bf*)(bbase + kc * 4096 + 512);    // tile nt0+1
        acc0 = __builtin_amdgcn_wmma_f32_16x16x32_bf16(false, A.v, false, B0,
                                                       (short)0, acc0, false, false);
        acc1 = __builtin_amdgcn_wmma_f32_16x16x32_bf16(false, A.v, false, B1,
                                                       (short)0, acc1, false, false);
    }
    const float* bias = (mat ? br : bl) + layer * HID;
    float* out = mat ? xr : xl;
    int col0  = nt0 * 16 + m;
    float bv0 = bias[col0];
    float bv1 = bias[col0 + 16];
    int rbase = rowTile * 16 + (hi ? 8 : 0);
    if (rbase + 7 < N) {                         // full tile: straight-line stores
        #pragma unroll
        for (int r = 0; r < 8; ++r) {
            float* rowp = out + (size_t)(rbase + r) * HID;
            rowp[col0]      = acc0[r] + bv0;
            rowp[col0 + 16] = acc1[r] + bv1;
        }
    } else {                                     // cold remainder path
        #pragma unroll
        for (int r = 0; r < 8; ++r) {
            if (rbase + r < N) {
                float* rowp = out + (size_t)(rbase + r) * HID;
                rowp[col0]      = acc0[r] + bv0;
                rowp[col0 + 16] = acc1[r] + bv1;
            }
        }
    }
}

// ---------------------------------------------------------------------------
__global__ void k_init(float* __restrict__ agg, unsigned* __restrict__ smaxkey,
                       float* __restrict__ denom, int N) {
    int i = blockIdx.x * blockDim.x + threadIdx.x;
    if (i < N * HID) agg[i] = 0.0f;
    if (i < N * HEADS) { smaxkey[i] = 0u; denom[i] = 0.0f; }
}

// ---------------------------------------------------------------------------
// Edge pass A: s[e,h] = sum_c lrelu(xl[src]+xr[dst]+ee)*att ; atomic seg-max
// ---------------------------------------------------------------------------
__global__ void k_edgeA(const int* __restrict__ ei, const float* __restrict__ ea,
                        const float* __restrict__ xl, const float* __restrict__ xr,
                        const float* __restrict__ We, const float* __restrict__ att,
                        float* __restrict__ sbuf, unsigned* __restrict__ smaxkey,
                        int E, int layer) {
    __shared__ float sWe[4 * HID];
    __shared__ float sAtt[HID];
    for (int i = threadIdx.x; i < 4 * HID; i += blockDim.x) sWe[i] = We[layer * 4 * HID + i];
    if (threadIdx.x < HID) sAtt[threadIdx.x] = att[layer * HID + threadIdx.x];
    __syncthreads();
    int idx = blockIdx.x * blockDim.x + threadIdx.x;
    if (idx >= E * HEADS) return;
    int e = idx >> 2, hd = idx & 3;
    int src = ei[e], dst = ei[E + e];
    float4 a4 = *(const float4*)(ea + (size_t)e * 4);
    const float4* pl = (const float4*)(xl + (size_t)src * HID + hd * CH);
    const float4* pr = (const float4*)(xr + (size_t)dst * HID + hd * CH);
    float s = 0.0f;
    #pragma unroll
    for (int q = 0; q < 8; ++q) {
        float4 va = pl[q], vb = pr[q];
        #pragma unroll
        for (int c2 = 0; c2 < 4; ++c2) {
            int col = hd * CH + q * 4 + c2;
            float m = (&va.x)[c2] + (&vb.x)[c2]
                    + a4.x * sWe[col]           + a4.y * sWe[HID + col]
                    + a4.z * sWe[2 * HID + col] + a4.w * sWe[3 * HID + col];
            float lr = (m > 0.0f) ? m : 0.2f * m;
            s = fmaf(lr, sAtt[col], s);
        }
    }
    sbuf[idx] = s;
    atomicMax(&smaxkey[dst * HEADS + hd], fkey(s));
}

// ---------------------------------------------------------------------------
// Edge pass B: es = exp(s - smax[dst]); denom[dst] += es
// ---------------------------------------------------------------------------
__global__ void k_edgeB(const int* __restrict__ ei, float* __restrict__ sbuf,
                        const unsigned* __restrict__ smaxkey, float* __restrict__ denom,
                        int E) {
    int idx = blockIdx.x * blockDim.x + threadIdx.x;
    if (idx >= E * HEADS) return;
    int e = idx >> 2, hd = idx & 3;
    int dst = ei[E + e];
    float mx = funkey(smaxkey[dst * HEADS + hd]);
    float es = __expf(sbuf[idx] - mx);
    sbuf[idx] = es;
    unsafeAtomicAdd(&denom[dst * HEADS + hd], es);   // global_atomic_add_f32
}

// ---------------------------------------------------------------------------
// Edge pass C: agg[dst] += (es/denom) * xl[src]   (32 fp32 atomics / thread)
// ---------------------------------------------------------------------------
__global__ void k_edgeC(const int* __restrict__ ei, const float* __restrict__ sbuf,
                        const float* __restrict__ denom, const float* __restrict__ xl,
                        float* __restrict__ agg, int E) {
    int idx = blockIdx.x * blockDim.x + threadIdx.x;
    if (idx >= E * HEADS) return;
    int e = idx >> 2, hd = idx & 3;
    int src = ei[e], dst = ei[E + e];
    float alpha = sbuf[idx] / (denom[dst * HEADS + hd] + 1e-16f);
    const float4* pl = (const float4*)(xl + (size_t)src * HID + hd * CH);
    float* pa = agg + (size_t)dst * HID + hd * CH;
    #pragma unroll
    for (int q = 0; q < 8; ++q) {
        float4 v = pl[q];
        unsafeAtomicAdd(&pa[q * 4 + 0], alpha * v.x);
        unsafeAtomicAdd(&pa[q * 4 + 1], alpha * v.y);
        unsafeAtomicAdd(&pa[q * 4 + 2], alpha * v.z);
        unsafeAtomicAdd(&pa[q * 4 + 3], alpha * v.w);
    }
}

// ---------------------------------------------------------------------------
// Node update: h = GELU(LN(gamma*(agg+cb)+beta)) + h ; refresh bf16 copy
// ---------------------------------------------------------------------------
__global__ void k_update(const float* __restrict__ agg, const float* __restrict__ cb,
                         const float* __restrict__ gamma, const float* __restrict__ beta,
                         const float* __restrict__ lng, const float* __restrict__ lnb,
                         float* __restrict__ h, __bf16* __restrict__ hbf, int layer) {
    int n = blockIdx.x, t = threadIdx.x;
    __shared__ float red[HID];
    float hc = agg[n * HID + t] + cb[layer * HID + t];
    float v  = gamma[n * HID + t] * hc + beta[n * HID + t];
    red[t] = v; __syncthreads();
    for (int s2 = 64; s2 > 0; s2 >>= 1) { if (t < s2) red[t] += red[t + s2]; __syncthreads(); }
    float mu = red[0] * (1.0f / HID); __syncthreads();
    float d = v - mu;
    red[t] = d * d; __syncthreads();
    for (int s2 = 64; s2 > 0; s2 >>= 1) { if (t < s2) red[t] += red[t + s2]; __syncthreads(); }
    float var = red[0] * (1.0f / HID);
    float y = gelu_erf(d * rsqrtf(var + 1e-5f) * lng[layer * HID + t] + lnb[layer * HID + t])
            + h[n * HID + t];
    h[n * HID + t]   = y;
    hbf[n * HID + t] = (__bf16)y;
}

// ---------------------------------------------------------------------------
// Decoder: delta = (GELU(h@W1+b1)@W2+b2)*(!fixed); out = [new_coords | delta]
// ---------------------------------------------------------------------------
__global__ void k_decoder(const float* __restrict__ h,
                          const float* __restrict__ W1, const float* __restrict__ b1,
                          const float* __restrict__ W2, const float* __restrict__ b2,
                          const float* __restrict__ x, const unsigned char* __restrict__ fixedm,
                          float* __restrict__ out, int N) {
    int n = blockIdx.x, t = threadIdx.x;   // 64 threads
    __shared__ float t1[64];
    float v = b1[t];
    for (int j = 0; j < HID; ++j) v = fmaf(h[n * HID + j], W1[j * 64 + t], v);
    t1[t] = gelu_erf(v);
    __syncthreads();
    if (t < 2) {
        float o = b2[t];
        #pragma unroll 8
        for (int j = 0; j < 64; ++j) o = fmaf(t1[j], W2[j * 2 + t], o);
        float dl = fixedm[n] ? 0.0f : o;
        out[(size_t)n * 2 + t] = x[n * 6 + t] + dl;          // new_coords
        out[(size_t)N * 2 + (size_t)n * 2 + t] = dl;         // delta
    }
}

// ---------------------------------------------------------------------------
extern "C" void kernel_launch(void* const* d_in, const int* in_sizes, int n_in,
                              void* d_out, int out_size, void* d_ws, size_t ws_size,
                              hipStream_t stream) {
    const float* x       = (const float*)d_in[0];
    const int*   ei      = (const int*)d_in[1];
    const float* eattr   = (const float*)d_in[2];
    const float* target  = (const float*)d_in[3];
    const unsigned char* fixedm = (const unsigned char*)d_in[4];
    const float* enc_W   = (const float*)d_in[5];
    const float* enc_b   = (const float*)d_in[6];
    const float* enc_g   = (const float*)d_in[7];
    const float* enc_be  = (const float*)d_in[8];
    const float* film_W1 = (const float*)d_in[9];
    const float* film_b1 = (const float*)d_in[10];
    const float* film_W2 = (const float*)d_in[11];
    const float* film_b2 = (const float*)d_in[12];
    const float* Wl      = (const float*)d_in[13];
    const float* bl      = (const float*)d_in[14];
    const float* Wr      = (const float*)d_in[15];
    const float* br      = (const float*)d_in[16];
    const float* We      = (const float*)d_in[17];
    const float* att     = (const float*)d_in[18];
    const float* cb      = (const float*)d_in[19];
    const float* ln_g    = (const float*)d_in[20];
    const float* ln_b    = (const float*)d_in[21];
    const float* dec_W1  = (const float*)d_in[22];
    const float* dec_b1  = (const float*)d_in[23];
    const float* dec_W2  = (const float*)d_in[24];
    const float* dec_b2  = (const float*)d_in[25];

    const int N = in_sizes[0] / 6;
    const int E = in_sizes[2] / 4;
    const int rowTiles = (N + 15) / 16;
    const size_t padN = (size_t)rowTiles * 16;

    // workspace carve (256B-aligned chunks)
    char* p = (char*)d_ws;
    size_t off = 0;
    auto take = [&](size_t bytes) -> void* {
        void* r = p + off;
        off += (bytes + 255) & ~(size_t)255;
        return r;
    };
    float*    h       = (float*)   take(padN * HID * 4);
    __bf16*   hbf     = (__bf16*)  take(padN * HID * 2);
    float*    gamma   = (float*)   take((size_t)N * HID * 4);
    float*    beta    = (float*)   take((size_t)N * HID * 4);
    float*    xlb     = (float*)   take(padN * HID * 4);
    float*    xrb     = (float*)   take(padN * HID * 4);
    float*    agg     = (float*)   take((size_t)N * HID * 4);
    float*    sbuf    = (float*)   take((size_t)E * HEADS * 4);
    unsigned* smaxkey = (unsigned*)take((size_t)N * HEADS * 4);
    float*    denom   = (float*)   take((size_t)N * HEADS * 4);
    __bf16*   wpack   = (__bf16*)  take((size_t)LAY * 2 * HID * HID * 2);

    // 1) repack weights to bf16 WMMA layout
    k_prep_w<<<(LAY * 2 * HID * HID) / 256, 256, 0, stream>>>(Wl, Wr, wpack);
    // 2) node encoder
    k_encoder<<<N, 128, 0, stream>>>(x, enc_W, enc_b, enc_g, enc_be, h, hbf);
    // 3) FiLM gamma/beta
    k_film<<<N, 256, 0, stream>>>(target, film_W1, film_b1, film_W2, film_b2, gamma, beta);

    const int nodeGrid = (N * HID + 255) / 256;
    const int edgeGrid = (E * HEADS + 255) / 256;
    const int gemmGrid = rowTiles * 2;   // 8 waves per strip, 4 waves (128 thr) per block

    for (int l = 0; l < LAY; ++l) {
        k_gemm_xlxr<<<gemmGrid, 128, 0, stream>>>(hbf, wpack, bl, br, xlb, xrb, l, N);
        k_init<<<nodeGrid, 256, 0, stream>>>(agg, smaxkey, denom, N);
        k_edgeA<<<edgeGrid, 256, 0, stream>>>(ei, eattr, xlb, xrb, We, att, sbuf, smaxkey, E, l);
        k_edgeB<<<edgeGrid, 256, 0, stream>>>(ei, sbuf, smaxkey, denom, E);
        k_edgeC<<<edgeGrid, 256, 0, stream>>>(ei, sbuf, denom, xlb, agg, E);
        k_update<<<N, 128, 0, stream>>>(agg, cb, gamma, beta, ln_g, ln_b, h, hbf, l);
    }
    // 4) decoder + mask + output assembly
    k_decoder<<<N, 64, 0, stream>>>(h, dec_W1, dec_b1, dec_W2, dec_b2, x, fixedm,
                                    (float*)d_out, N);
}
